// W8A8OF16LinearStaticScale_4861902979222
// MI455X (gfx1250) — 8-bit matrix operand order repaired and checked by reading, compiled, not run on hardware
//
#include <hip/hip_runtime.h>
#include <stdint.h>

typedef int v8i __attribute__((ext_vector_type(8)));

#define TOKENS 8192
#define IN_F   4096
#define OUT_F  4096

#define BM 128
#define BN 128
#define BK 128
#define KT (IN_F / BK)
#define LDS_STRIDE 144
#define TILE_BYTES (BM * LDS_STRIDE)
#define BUF_BYTES  (2 * TILE_BYTES)

#ifndef USE_ASYNC_LDS
#define USE_ASYNC_LDS 1
#endif

template <int OFF>
__device__ __forceinline__ void async_ld_b128_off(unsigned lds_off, const void* gptr) {
    asm volatile("global_load_async_to_lds_b128 %0, %1, off offset:%2"
                 :: "v"(lds_off), "v"((unsigned long long)(uintptr_t)gptr), "i"(OFF)
                 : "memory");
}

__device__ __forceinline__ void stage_async(unsigned xlds, const int8_t* xg,
                                            unsigned wlds, const int8_t* wg) {
    async_ld_b128_off<0 >(xlds, xg);
    async_ld_b128_off<16>(xlds, xg);
    async_ld_b128_off<32>(xlds, xg);
    async_ld_b128_off<48>(xlds, xg);
    async_ld_b128_off<0 >(wlds, wg);
    async_ld_b128_off<16>(wlds, wg);
    async_ld_b128_off<32>(wlds, wg);
    async_ld_b128_off<48>(wlds, wg);
}

__global__ __launch_bounds__(256)
void w8a8_of16_gemm_kernel(const int8_t* __restrict__ X,
                           const int8_t* __restrict__ W,
                           const float*  __restrict__ scale,
                           const _Float16* __restrict__ bias,
                           _Float16* __restrict__ out)
{
    __shared__ __align__(16) unsigned char smem[2 * BUF_BYTES];

    const int tid  = threadIdx.x;
    const int lane = tid & 31;
    const int wave = tid >> 5;
    const int wm   = wave >> 1;
    const int wn   = wave & 1;

    const int m0 = blockIdx.y * BM;
    const int n0 = blockIdx.x * BN;

    const int sRow = tid >> 1;
    const int sKb  = (tid & 1) * 64;

    const int8_t* xg = X + (size_t)(m0 + sRow) * IN_F + sKb;
    const int8_t* wg = W + (size_t)(n0 + sRow) * IN_F + sKb;
    const unsigned sOff      = (unsigned)(sRow * LDS_STRIDE + sKb);
    const unsigned smem_base = (unsigned)(uintptr_t)&smem[0];
    const unsigned xlds0 = smem_base + sOff;
    const unsigned wlds0 = smem_base + TILE_BYTES + sOff;

    v8i acc[2][4];
#pragma unroll
    for (int ti = 0; ti < 2; ++ti)
#pragma unroll
        for (int tj = 0; tj < 4; ++tj)
#pragma unroll
            for (int r = 0; r < 8; ++r) acc[ti][tj][r] = 0;

    const int aRow = lane & 15;
    const int aKb  = (lane >> 4) * 16;
    const int bCol = lane & 15;
    const int bKb  = (lane >> 4) * 16;

#if USE_ASYNC_LDS
    stage_async(xlds0, xg, wlds0, wg);
#endif

#pragma unroll 1
    for (int kt = 0; kt < KT; ++kt) {
        const int buf = kt & 1;

#if USE_ASYNC_LDS
        if (kt + 1 < KT) {
            const unsigned nb = (unsigned)((kt + 1) & 1) * BUF_BYTES;
            const size_t ko = (size_t)(kt + 1) * BK;
            stage_async(xlds0 + nb, xg + ko, wlds0 + nb, wg + ko);
            asm volatile("s_wait_asynccnt 0x8" ::: "memory");
        } else {
            asm volatile("s_wait_asynccnt 0x0" ::: "memory");
        }
        __syncthreads();
#else
        {
            const unsigned b = (unsigned)buf * BUF_BYTES;
            const size_t ko = (size_t)kt * BK;
#pragma unroll
            for (int j = 0; j < 4; ++j) {
                int4 ax = *(const int4*)(xg + ko + j * 16);
                int4 aw = *(const int4*)(wg + ko + j * 16);
                *(int4*)(&smem[b + sOff + j * 16])              = ax;
                *(int4*)(&smem[b + TILE_BYTES + sOff + j * 16]) = aw;
            }
        }
        __syncthreads();
#endif

        const unsigned char* xb = &smem[(unsigned)buf * BUF_BYTES];
        const unsigned char* wb = xb + TILE_BYTES;

#pragma unroll
        for (int kh = 0; kh < 2; ++kh) {
            const int khb = kh * 64;

            v8i A[2];
#pragma unroll
            for (int ti = 0; ti < 2; ++ti) {
                const unsigned char* p =
                    xb + (wm * 32 + ti * 16 + aRow) * LDS_STRIDE + khb + aKb;
                int2 q0 = *(const int2*)(p);
                int2 q1 = *(const int2*)(p + 8);
                int2 q2 = *(const int2*)(p + 32);
                int2 q3 = *(const int2*)(p + 40);
                A[ti][0] = q0.x; A[ti][1] = q0.y;
                A[ti][2] = q1.x; A[ti][3] = q1.y;
                A[ti][4] = q2.x; A[ti][5] = q2.y;
                A[ti][6] = q3.x; A[ti][7] = q3.y;
            }
            v8i B[4];
#pragma unroll
            for (int tj = 0; tj < 4; ++tj) {
                const unsigned char* p =
                    wb + (wn * 64 + tj * 16 + bCol) * LDS_STRIDE + khb + bKb;
                int4 q0 = *(const int4*)(p);
                int4 q1 = *(const int4*)(p + 32);
                B[tj][0] = q0.x; B[tj][1] = q0.y; B[tj][2] = q0.z; B[tj][3] = q0.w;
                B[tj][4] = q1.x; B[tj][5] = q1.y; B[tj][6] = q1.z; B[tj][7] = q1.w;
            }

#pragma unroll
            for (int ti = 0; ti < 2; ++ti)
#pragma unroll
                for (int tj = 0; tj < 4; ++tj)
                    acc[ti][tj] = __builtin_amdgcn_wmma_i32_16x16x64_iu8(
                         true, A[ti],  true, B[tj],
                        acc[ti][tj],  false,  false);
        }

        __syncthreads();
    }

    const int mBase   = m0 + wm * 32;
    const int nBase   = n0 + wn * 64;
    const int rowHalf = (lane >> 4) * 8;
    const int colLane = lane & 15;

#pragma unroll
    for (int tj = 0; tj < 4; ++tj) {
        const int n = nBase + tj * 16 + colLane;
        const float s = scale[n];
        const float b = (float)bias[n];
#pragma unroll
        for (int ti = 0; ti < 2; ++ti) {
#pragma unroll
            for (int r = 0; r < 8; ++r) {
                const int m = mBase + ti * 16 + rowHalf + r;
                out[(size_t)m * OUT_F + n] =
                    (_Float16)((float)acc[ti][tj][r] * s + b);
            }
        }
    }
}

__global__ void bias_copy_kernel(const _Float16* __restrict__ bias,
                                 _Float16* __restrict__ out_bias)
{
    int i = blockIdx.x * blockDim.x + threadIdx.x;
    if (i < OUT_F) out_bias[i] = bias[i];
}

extern "C" void kernel_launch(void* const* d_in, const int* in_sizes, int n_in,
                              void* d_out, int out_size, void* d_ws, size_t ws_size,
                              hipStream_t stream) {
    (void)in_sizes; (void)n_in; (void)d_ws; (void)ws_size; (void)out_size;
    const int8_t*   X     = (const int8_t*)d_in[0];
    const int8_t*   Wt    = (const int8_t*)d_in[1];
    const float*    scale = (const float*)d_in[2];
    const _Float16* bias  = (const _Float16*)d_in[3];
    _Float16*       out   = (_Float16*)d_out;

    dim3 grid(OUT_F / BN, TOKENS / BM);
    w8a8_of16_gemm_kernel<<<grid, 256, 0, stream>>>(X, Wt, scale, bias, out);
    bias_copy_kernel<<<OUT_F / 256, 256, 0, stream>>>(bias, out + (size_t)TOKENS * OUT_F);
}
